// GroupBlock_49039936586282
// MI455X (gfx1250) — compile-verified, matrix operand order checked by reading, not run on hardware
//
#include <hip/hip_runtime.h>
#include <hip/hip_bf16.h>
#include <math.h>
#include <stdint.h>

typedef __bf16 bf16;
typedef __attribute__((ext_vector_type(16))) __bf16 v16bf;
typedef __attribute__((ext_vector_type(8)))  __bf16 v8bf;
typedef __attribute__((ext_vector_type(8)))  float  v8f;

static constexpr int DIMC   = 384;
static constexpr int HEADS  = 12;
static constexpr int HD     = 32;
static constexpr int WSZ    = 8;
static constexpr int PTOK   = 64;            // tokens per window
static constexpr int MLPH   = 1536;
static constexpr int BATCH  = 16;
static constexpr int WIMG   = 64;
static constexpr int MTOT   = BATCH * 4096;  // 65536 rows total
static constexpr int QKVN   = 3 * DIMC;      // 1152

// ---------- token index remapping (orig <-> window-grouped order) ----------
__device__ __forceinline__ int win_to_orig(int r) {
  int b = r >> 12; int rem = r & 4095;
  int g = rem >> 6; int p = rem & 63;
  int hg = g >> 3, wg = g & 7;
  int ph = p >> 3, pw = p & 7;
  int hi = hg * WSZ + ph, wi = wg * WSZ + pw;
  return (b << 12) + hi * WIMG + wi;
}
__device__ __forceinline__ int orig_to_win(int t) {
  int b = t >> 12; int n = t & 4095;
  int hi = n >> 6, wi = n & 63;
  int g = (hi >> 3) * 8 + (wi >> 3);
  int p = (hi & 7) * 8 + (wi & 7);
  return (b << 12) + g * PTOK + p;
}

// ---------- WMMA 16x32 bf16 fragment load (A layout; B uses same layout on B^T) ----
// Per ISA 16-bit A-matrix table: lane half=0 holds K 0..7 & 16..23, half=1 holds
// K 8..15 & 24..31, row m = lane&15.  Two 16-byte loads per lane.
__device__ __forceinline__ v16bf load_frag(const bf16* tile, int ld) {
  int lane = threadIdx.x & 31;
  int m = lane & 15, half = lane >> 4;
  const bf16* p = tile + (size_t)m * ld + half * 8;
  union { v16bf v; v8bf h[2]; } u;
  u.h[0] = *(const v8bf*)(p);
  u.h[1] = *(const v8bf*)(p + 16);
  return u.v;
}

// ---------- async global->LDS staging of a rows x 32 bf16 panel ----------
// Each 16B chunk = 8 bf16.  chunks = rows*4, must be a multiple of nthreads so
// every lane issues the same number of async ops (uniform ASYNCcnt, full EXEC).
__device__ __forceinline__ void stage_panel_async(const bf16* __restrict__ src, int ldk,
                                                  bf16* dst_lds, int rows,
                                                  int tid, int nthreads) {
  int nchunks = rows * 4;
  for (int c = tid; c < nchunks; c += nthreads) {
    int r = c >> 2, q = c & 3;
    const bf16* g = src + (size_t)r * ldk + q * 8;
    uint32_t loff = (uint32_t)(uintptr_t)(dst_lds + r * 32 + q * 8);
    asm volatile("global_load_async_to_lds_b128 %0, %1, off"
                 :: "v"(loff), "v"(g) : "memory");
  }
}

// ---------- weight f32 -> bf16 ----------
__global__ void cvt_f32_bf16(const float* __restrict__ in, bf16* __restrict__ out, int n) {
  int i = blockIdx.x * 256 + threadIdx.x;
  if (i < n) out[i] = (bf16)in[i];
}

// ---------- layernorm (+optional orig->window reorder), f32 in, bf16 out -------
__global__ __launch_bounds__(128)
void ln_kernel(const float* __restrict__ x, const float* __restrict__ g,
               const float* __restrict__ bta, bf16* __restrict__ out, int reorder) {
  int t = blockIdx.x;
  int tid = threadIdx.x;  // 128 threads, 3 elems each
  const float* row = x + (size_t)t * DIMC;
  float v0 = row[tid], v1 = row[tid + 128], v2 = row[tid + 256];
  __shared__ float rs[128], rq[128];
  rs[tid] = v0 + v1 + v2;
  rq[tid] = v0 * v0 + v1 * v1 + v2 * v2;
  __syncthreads();
  for (int s = 64; s > 0; s >>= 1) {
    if (tid < s) { rs[tid] += rs[tid + s]; rq[tid] += rq[tid + s]; }
    __syncthreads();
  }
  float mean = rs[0] * (1.0f / DIMC);
  float var  = rq[0] * (1.0f / DIMC) - mean * mean;
  float inv  = rsqrtf(var + 1e-5f);
  int to = reorder ? orig_to_win(t) : t;
  bf16* orow = out + (size_t)to * DIMC;
  orow[tid]       = (bf16)((v0 - mean) * inv * g[tid]       + bta[tid]);
  orow[tid + 128] = (bf16)((v1 - mean) * inv * g[tid + 128] + bta[tid + 128]);
  orow[tid + 256] = (bf16)((v2 - mean) * inv * g[tid + 256] + bta[tid + 256]);
}

// ---------- WMMA GEMM with async-LDS double-buffered staging ----------
// C[M,N] = A[M,K] @ W[N,K]^T.  Block = 256 threads (8 waves), tile 128 x 64.
// Per k-step (32): stage A panel 128x32 + B panel 64x32 into LDS with
// global_load_async_to_lds_b128 (3 async instrs per wave per stage), double
// buffered; each wave computes 16 rows x 64 cols = 4 WMMAs from ds_load_b128
// fragments.
enum { EPI_QKV = 0, EPI_PROJ = 1, EPI_FC1 = 2, EPI_FC2 = 3 };

template <int EPI>
__global__ __launch_bounds__(256)
void gemm_wmma(const bf16* __restrict__ A, const bf16* __restrict__ W,
               int K, int N,
               const float* __restrict__ bias, const float* __restrict__ resid,
               float* __restrict__ outF, bf16* __restrict__ outB) {
  __shared__ __align__(16) bf16 sA[2][128][32];  // 16 KB
  __shared__ __align__(16) bf16 sB[2][64][32];   //  8 KB

  int tid  = threadIdx.x;
  int wave = tid >> 5;
  int lane = tid & 31;
  int rowblk = blockIdx.y * 128;
  int r0 = rowblk + wave * 16;
  int c0 = blockIdx.x * 64;

  const bf16* Ab = A + (size_t)rowblk * K;
  const bf16* Wb = W + (size_t)c0 * K;
  const int nsteps = K / 32;

  // prologue: stage k-step 0 into buffer 0
  stage_panel_async(Ab, K, &sA[0][0][0], 128, tid, 256);
  stage_panel_async(Wb, K, &sB[0][0][0], 64, tid, 256);

  v8f acc[4] = {};
  for (int i = 0; i < nsteps; ++i) {
    int cur = i & 1;
    if (i + 1 < nsteps) {
      int nxt = cur ^ 1;
      stage_panel_async(Ab + (i + 1) * 32, K, &sA[nxt][0][0], 128, tid, 256);
      stage_panel_async(Wb + (i + 1) * 32, K, &sB[nxt][0][0], 64, tid, 256);
      asm volatile("s_wait_asynccnt 0x3" ::: "memory");  // stage i done, i+1 in flight
    } else {
      asm volatile("s_wait_asynccnt 0x0" ::: "memory");
    }
    __syncthreads();  // all waves' stage-i data visible

    v16bf a = load_frag(&sA[cur][wave * 16][0], 32);
#pragma unroll
    for (int j = 0; j < 4; j++) {
      v16bf b = load_frag(&sB[cur][16 * j][0], 32);
      acc[j] = __builtin_amdgcn_wmma_f32_16x16x32_bf16(
          false, a, false, b, (short)0, acc[j], false, false);
    }
    __syncthreads();  // done reading buf[cur] before it is restaged
  }

  int nloc = lane & 15, half = lane >> 4;
#pragma unroll
  for (int j = 0; j < 4; j++) {
    int c = c0 + 16 * j + nloc;
#pragma unroll
    for (int v = 0; v < 8; v++) {
      int r = r0 + v + 8 * half;
      float val = acc[j][v];
      if (EPI == EPI_QKV) {
        outB[(size_t)r * N + c] = (bf16)val;
      } else if (EPI == EPI_PROJ) {
        int ro = win_to_orig(r);
        outF[(size_t)ro * N + c] = resid[(size_t)ro * N + c] + val + bias[c];
      } else if (EPI == EPI_FC1) {
        float z = val + bias[c];
        float gl = 0.5f * z * (1.0f + erff(z * 0.70710678118f));
        outB[(size_t)r * N + c] = (bf16)gl;
      } else {  // EPI_FC2
        outF[(size_t)r * N + c] = resid[(size_t)r * N + c] + val + bias[c];
      }
    }
  }
}

// ---------- windowed attention: one wave per (window, head) ----------
__global__ __launch_bounds__(32)
void attn_wmma(const bf16* __restrict__ qkv, bf16* __restrict__ o) {
  __shared__ __align__(16) float sS[PTOK][PTOK];   // 16 KB scores (f32)
  __shared__ __align__(16) bf16  sPm[PTOK][PTOK];  //  8 KB probs (bf16)
  __shared__ __align__(16) bf16  sVt[HD][PTOK];    //  4 KB V^T (bf16)

  int wid = blockIdx.x;
  int h = wid % HEADS;
  int w = wid / HEADS;      // window 0..1023
  int base = w * PTOK;      // row in window-ordered token space
  const bf16* Q  = qkv + (size_t)base * QKVN + h * HD;
  const bf16* Kp = Q + DIMC;
  const bf16* Vp = Q + 2 * DIMC;

  int lane = threadIdx.x;
  int nloc = lane & 15, half = lane >> 4;
  const float scale = 0.1767766953f;  // 1/sqrt(32)

  // S = Q @ K^T  (4x4 tiles, one 16x16x32 WMMA each since hd==32)
#pragma unroll
  for (int mi = 0; mi < 4; mi++) {
    v16bf a = load_frag(Q + (size_t)(16 * mi) * QKVN, QKVN);
#pragma unroll
    for (int ni = 0; ni < 4; ni++) {
      v16bf b = load_frag(Kp + (size_t)(16 * ni) * QKVN, QKVN);
      v8f c = {};
      c = __builtin_amdgcn_wmma_f32_16x16x32_bf16(false, a, false, b, (short)0, c,
                                                  false, false);
#pragma unroll
      for (int v = 0; v < 8; v++)
        sS[16 * mi + v + 8 * half][16 * ni + nloc] = c[v] * scale;
    }
  }

  // stage V^T into LDS for B-fragments of P@V
  for (int idx = lane; idx < PTOK * HD; idx += 32) {
    int r = idx >> 5;   // token
    int c = idx & 31;   // hd channel
    sVt[c][r] = Vp[(size_t)r * QKVN + c];
  }
  __syncthreads();

  // softmax over rows (2 rows per lane)
  for (int rr = 0; rr < 2; rr++) {
    int r = lane + 32 * rr;
    float mx = -1e30f;
    for (int c = 0; c < PTOK; c++) mx = fmaxf(mx, sS[r][c]);
    float sum = 0.f;
    for (int c = 0; c < PTOK; c++) { float e = expf(sS[r][c] - mx); sS[r][c] = e; sum += e; }
    float inv = 1.0f / sum;
    for (int c = 0; c < PTOK; c++) sPm[r][c] = (bf16)(sS[r][c] * inv);
  }
  __syncthreads();

  // O = P @ V : 4x2 tiles, K=64 -> two k-steps
#pragma unroll
  for (int mi = 0; mi < 4; mi++) {
#pragma unroll
    for (int nj = 0; nj < 2; nj++) {
      v8f c = {};
#pragma unroll
      for (int k2 = 0; k2 < 2; k2++) {
        v16bf a = load_frag(&sPm[16 * mi][32 * k2], PTOK);
        v16bf b = load_frag(&sVt[16 * nj][32 * k2], PTOK);
        c = __builtin_amdgcn_wmma_f32_16x16x32_bf16(false, a, false, b, (short)0, c,
                                                    false, false);
      }
#pragma unroll
      for (int v = 0; v < 8; v++) {
        int r = base + 16 * mi + v + 8 * half;
        int col = h * HD + 16 * nj + nloc;
        o[(size_t)r * DIMC + col] = (bf16)c[v];
      }
    }
  }
}

// =============================== host launch ===============================
extern "C" void kernel_launch(void* const* d_in, const int* in_sizes, int n_in,
                              void* d_out, int out_size, void* d_ws, size_t ws_size,
                              hipStream_t stream) {
  const float* x      = (const float*)d_in[0];
  const float* qkv_w  = (const float*)d_in[3];
  const float* proj_w = (const float*)d_in[4];
  const float* proj_b = (const float*)d_in[5];
  const float* ln1_g  = (const float*)d_in[6];
  const float* ln1_b  = (const float*)d_in[7];
  const float* ln2_g  = (const float*)d_in[8];
  const float* ln2_b  = (const float*)d_in[9];
  const float* fc1_w  = (const float*)d_in[10];
  const float* fc1_b  = (const float*)d_in[11];
  const float* fc2_w  = (const float*)d_in[12];
  const float* fc2_b  = (const float*)d_in[13];
  float* out = (float*)d_out;

  // ---- workspace layout (256B aligned) ----
  char* p = (char*)d_ws;
  size_t off = 0;
  auto take = [&](size_t bytes) {
    void* r = p + off;
    off = (off + bytes + 255) & ~(size_t)255;
    return r;
  };
  bf16* w_qkv  = (bf16*)take((size_t)QKVN * DIMC * 2);
  bf16* w_proj = (bf16*)take((size_t)DIMC * DIMC * 2);
  bf16* w_fc1  = (bf16*)take((size_t)MLPH * DIMC * 2);
  bf16* w_fc2  = (bf16*)take((size_t)DIMC * MLPH * 2);
  bf16* xn_bf  = (bf16*)take((size_t)MTOT * DIMC * 2);   // LN out (reused for LN2)
  float* x1    = (float*)take((size_t)MTOT * DIMC * 4);  // post-attention residual
  // region: qkv (151MB) + attn (50MB); h1 (201MB) aliases the whole region
  bf16* qkv_bf = (bf16*)take((size_t)MTOT * QKVN * 2);
  bf16* attn_bf= (bf16*)take((size_t)MTOT * DIMC * 2);
  bf16* h1     = qkv_bf;  // alias: FC1 output written after qkv/attn are dead

  // 1) weights -> bf16
  auto cvt = [&](const float* src, bf16* dst, int n) {
    cvt_f32_bf16<<<(n + 255) / 256, 256, 0, stream>>>(src, dst, n);
  };
  cvt(qkv_w,  w_qkv,  QKVN * DIMC);
  cvt(proj_w, w_proj, DIMC * DIMC);
  cvt(fc1_w,  w_fc1,  MLPH * DIMC);
  cvt(fc2_w,  w_fc2,  DIMC * MLPH);

  // 2) LN1 + window reorder -> bf16
  ln_kernel<<<MTOT, 128, 0, stream>>>(x, ln1_g, ln1_b, xn_bf, 1);

  // 3) QKV GEMM: [65536,384] x [1152,384]^T -> bf16 [65536,1152]
  gemm_wmma<EPI_QKV><<<dim3(QKVN / 64, MTOT / 128), 256, 0, stream>>>(
      xn_bf, w_qkv, DIMC, QKVN, nullptr, nullptr, nullptr, qkv_bf);

  // 4) windowed attention (1024 windows x 12 heads), one wave each
  attn_wmma<<<(MTOT / PTOK) * HEADS, 32, 0, stream>>>(qkv_bf, attn_bf);

  // 5) proj GEMM + bias + residual + window->orig reorder -> f32 x1
  gemm_wmma<EPI_PROJ><<<dim3(DIMC / 64, MTOT / 128), 256, 0, stream>>>(
      attn_bf, w_proj, DIMC, DIMC, proj_b, x, x1, nullptr);

  // 6) LN2 -> bf16 (orig order, reuse xn_bf)
  ln_kernel<<<MTOT, 128, 0, stream>>>(x1, ln2_g, ln2_b, xn_bf, 0);

  // 7) FC1 GEMM + bias + exact GELU -> bf16 h1
  gemm_wmma<EPI_FC1><<<dim3(MLPH / 64, MTOT / 128), 256, 0, stream>>>(
      xn_bf, w_fc1, DIMC, MLPH, fc1_b, nullptr, nullptr, h1);

  // 8) FC2 GEMM + bias + residual -> f32 d_out
  gemm_wmma<EPI_FC2><<<dim3(DIMC / 64, MTOT / 128), 256, 0, stream>>>(
      h1, w_fc2, MLPH, DIMC, fc2_b, x1, out, nullptr);
}
